// SparseECMoeBlock_43121471652483
// MI455X (gfx1250) — matrix operand order repaired and checked by reading, not compiled, not run on hardware
//
#include <hip/hip_runtime.h>
#include <hip/hip_bf16.h>
#include <math.h>

// ---------------- problem dims (static) ----------------
#define NTOK   (8 * 4096)   // B*L = 32768
#define DMODEL 768
#define NEXP   16
#define CAP    4096         // ceil(N/E * top_k)
#define FEXP   3072
#define FSH    1536
#define NBINS  2048

typedef __attribute__((ext_vector_type(16))) __bf16 v16bf;
typedef __attribute__((ext_vector_type(8)))  __bf16 v8bf;
typedef __attribute__((ext_vector_type(8)))  float  v8f;

// ---------------- tiling ----------------
#define BM  128
#define BN  64
#define BK  32
#define LDH 40   // LDS row stride in halves (padded to dodge bank conflicts)

__device__ inline v8f zero_v8f() {
    v8f z;
#pragma unroll
    for (int i = 0; i < 8; ++i) z[i] = 0.f;
    return z;
}

// gfx1250 async copy: global -> LDS, 16B per lane, tracked by ASYNCcnt.
// INST_OFFSET applies to BOTH the LDS and the global address (ISA 08_async_tensor §4.4),
// so one base pair + offset:16 covers the second 16B chunk.
__device__ inline void async_copy_b128(unsigned lds_off, unsigned long long gaddr) {
    asm volatile("global_load_async_to_lds_b128 %0, %1, off"
                 :: "v"(lds_off), "v"(gaddr) : "memory");
    asm volatile("global_load_async_to_lds_b128 %0, %1, off offset:16"
                 :: "v"(lds_off), "v"(gaddr) : "memory");
}
__device__ inline void wait_asynccnt0() {
    asm volatile("s_wait_asynccnt 0x0" ::: "memory");
}

// A fragment (16x32 bf16): lane -> M = lane&15 ; K elems: e<8 -> e, e>=8 -> e+8,
// plus +8 for lanes 16..31  => two contiguous 8-half chunks at +b0 and +b0+16.
__device__ inline v16bf load_frag_a(const __bf16* As, int mbase, int lane) {
    int row = mbase + (lane & 15);
    int b0  = (lane >= 16) ? 8 : 0;
    const __bf16* p = As + row * LDH + b0;
    v8bf lo = *(const v8bf*)p;
    v8bf hi = *(const v8bf*)(p + 16);
    v16bf f;
#pragma unroll
    for (int i = 0; i < 8; ++i) { f[i] = lo[i]; f[i + 8] = hi[i]; }
    return f;
}

// B fragment (32x16 bf16): lane -> N = lane&15 ; K elems: e<8 -> e, e>=8 -> e+8,
// plus +8 for lanes 16..31  => two contiguous 8-half chunks at +b0 and +b0+16.
__device__ inline v16bf load_frag_b(const __bf16* Bs, int nbase, int lane) {
    int col = nbase + (lane & 15);
    int b0  = (lane >= 16) ? 8 : 0;
    const __bf16* p = Bs + col * LDH + b0;
    v8bf lo = *(const v8bf*)p;
    v8bf hi = *(const v8bf*)(p + 16);
    v16bf f;
#pragma unroll
    for (int i = 0; i < 8; ++i) { f[i] = lo[i]; f[i + 8] = hi[i]; }
    return f;
}

__device__ inline v8bf cvt8_f32_bf16(float4 a, float4 b) {
    v8bf r;
    r[0] = (__bf16)a.x; r[1] = (__bf16)a.y; r[2] = (__bf16)a.z; r[3] = (__bf16)a.w;
    r[4] = (__bf16)b.x; r[5] = (__bf16)b.y; r[6] = (__bf16)b.z; r[7] = (__bf16)b.w;
    return r;
}

// ================= gate: scores[E,N] = sigmoid(x @ gw^T), plus x -> bf16 copy ==========
__global__ __launch_bounds__(256)
void gate_kernel(const float* __restrict__ X, const float* __restrict__ GW,
                 float* __restrict__ scores, __bf16* __restrict__ Xb) {
    __shared__ float gws[NEXP * DMODEL];   // 48 KB
    int tid = threadIdx.x;
    for (int i = tid; i < NEXP * DMODEL; i += 256) gws[i] = GW[i];
    __syncthreads();

    int t = blockIdx.x * 256 + tid;
    float acc[NEXP];
#pragma unroll
    for (int e = 0; e < NEXP; ++e) acc[e] = 0.f;

    const float* xr = X + (size_t)t * DMODEL;
    __bf16* xb = Xb + (size_t)t * DMODEL;
    for (int d = 0; d < DMODEL; ++d) {
        float xv = xr[d];
        xb[d] = (__bf16)xv;
#pragma unroll
        for (int e = 0; e < NEXP; ++e) acc[e] = fmaf(xv, gws[e * DMODEL + d], acc[e]);
    }
#pragma unroll
    for (int e = 0; e < NEXP; ++e)
        scores[(size_t)e * NTOK + t] = 1.f / (1.f + __expf(-acc[e]));
}

// ================= expert-choice top-C: histogram select ==============================
__global__ void hist_kernel(const float* __restrict__ scores, int* __restrict__ hist) {
    int e = blockIdx.y;
    const float* s = scores + (size_t)e * NTOK;
    int* h = hist + e * NBINS;
    for (int t = blockIdx.x * blockDim.x + threadIdx.x; t < NTOK; t += gridDim.x * blockDim.x) {
        int b = (int)(s[t] * (float)NBINS);
        b = b < 0 ? 0 : (b > NBINS - 1 ? NBINS - 1 : b);
        atomicAdd(&h[b], 1);
    }
}

__global__ void thresh_kernel(const int* __restrict__ hist, int* __restrict__ thr) {
    if (threadIdx.x != 0) return;
    int e = blockIdx.x;
    const int* h = hist + e * NBINS;
    int cum = 0, T = 0, nHi = 0;
    for (int b = NBINS - 1; b >= 0; --b) {
        int c = h[b];
        if (cum + c >= CAP || b == 0) { T = b; nHi = cum; break; }
        cum += c;
    }
    thr[e * 3 + 0] = T;
    thr[e * 3 + 1] = nHi;
    thr[e * 3 + 2] = CAP - nHi;
}

__global__ void compact_kernel(const float* __restrict__ scores, const int* __restrict__ thr,
                               int* __restrict__ cntHi, int* __restrict__ cntEq,
                               int* __restrict__ ids, float* __restrict__ tsc) {
    int e = blockIdx.y;
    int T = thr[e * 3], nHi = thr[e * 3 + 1], r = thr[e * 3 + 2];
    const float* s = scores + (size_t)e * NTOK;
    for (int t = blockIdx.x * blockDim.x + threadIdx.x; t < NTOK; t += gridDim.x * blockDim.x) {
        float sv = s[t];
        int b = (int)(sv * (float)NBINS);
        b = b < 0 ? 0 : (b > NBINS - 1 ? NBINS - 1 : b);
        int pos = -1;
        if (b > T) {
            pos = atomicAdd(&cntHi[e], 1);
        } else if (b == T) {
            int p = atomicAdd(&cntEq[e], 1);
            if (p < r) pos = nHi + p;
        }
        if (pos >= 0) { ids[e * CAP + pos] = t; tsc[e * CAP + pos] = sv; }
    }
}

// ================= GEMM1: H = silu(Xg @ Wg^T) * (Xg @ Wu^T) ==========================
// A rows gathered via ids (or identity from `base`), copied global->LDS with the
// gfx1250 async-to-LDS path (no VGPR staging). Weights fp32 [F,K] converted to bf16
// in registers while staging into LDS. v_wmma_f32_16x16x32_bf16 core.
__global__ __launch_bounds__(256)
void gemm1_kernel(const __bf16* __restrict__ X, const int* __restrict__ ids, int base,
                  const float* __restrict__ Wg, const float* __restrict__ Wu,
                  __bf16* __restrict__ H, int F, int K) {
    __shared__ __bf16 As[BM * LDH];    // 10 KB
    __shared__ __bf16 Bgs[BN * LDH];   // 5 KB
    __shared__ __bf16 Bus[BN * LDH];   // 5 KB

    int tid  = threadIdx.x;
    int lane = tid & 31;
    int wid  = tid >> 5;
    int wm   = wid & 3;    // wave M offset = wm*32
    int wn   = wid >> 2;   // wave N offset = wn*32
    int blockN = blockIdx.x;
    int blockM = blockIdx.y;

    // A staging: 2 threads per row, 32B each, async global->LDS
    int arow  = tid >> 1;
    int ahalf = tid & 1;
    int slot  = blockM * BM + arow;
    int tokA  = ids ? ids[slot] : (base + slot);
    unsigned long long gA =
        (unsigned long long)(uintptr_t)(X + (size_t)tokA * K + ahalf * 16);
    unsigned ldsA = (unsigned)(uintptr_t)&As[arow * LDH + ahalf * 16]; // addr[31:0] = LDS offset

    // B staging: 4 threads per feature row, 8 floats each
    int brow = tid >> 2;
    int bq   = tid & 3;
    const float* gptr = Wg + (size_t)(blockN * BN + brow) * K + bq * 8;
    const float* uptr = Wu + (size_t)(blockN * BN + brow) * K + bq * 8;

    v8f accg[2][2], accu[2][2];
#pragma unroll
    for (int m = 0; m < 2; ++m)
#pragma unroll
        for (int n = 0; n < 2; ++n) { accg[m][n] = zero_v8f(); accu[m][n] = zero_v8f(); }

    for (int k0 = 0; k0 < K; k0 += BK) {
        // ---- stage A: async copy, no VGPR round-trip ----
        async_copy_b128(ldsA, gA + (unsigned long long)k0 * 2);

        // ---- stage B (fp32 -> bf16 in registers) ----
        float4 g0 = *(const float4*)(gptr + k0);
        float4 g1 = *(const float4*)(gptr + k0 + 4);
        float4 u0 = *(const float4*)(uptr + k0);
        float4 u1 = *(const float4*)(uptr + k0 + 4);
        *(v8bf*)&Bgs[brow * LDH + bq * 8] = cvt8_f32_bf16(g0, g1);
        *(v8bf*)&Bus[brow * LDH + bq * 8] = cvt8_f32_bf16(u0, u1);

        if (k0 + BK < K) {   // hint next weight K-tile toward the caches
            __builtin_prefetch(gptr + k0 + BK, 0, 1);
            __builtin_prefetch(uptr + k0 + BK, 0, 1);
        }
        wait_asynccnt0();
        __syncthreads();

        v16bf af[2], bgf[2], buf_[2];
#pragma unroll
        for (int m = 0; m < 2; ++m) af[m] = load_frag_a(As, wm * 32 + m * 16, lane);
#pragma unroll
        for (int n = 0; n < 2; ++n) {
            bgf[n]  = load_frag_b(Bgs, wn * 32 + n * 16, lane);
            buf_[n] = load_frag_b(Bus, wn * 32 + n * 16, lane);
        }
#pragma unroll
        for (int m = 0; m < 2; ++m)
#pragma unroll
            for (int n = 0; n < 2; ++n) {
                accg[m][n] = __builtin_amdgcn_wmma_f32_16x16x32_bf16(
                    false, af[m], false, bgf[n], (short)0, accg[m][n], false, false);
                accu[m][n] = __builtin_amdgcn_wmma_f32_16x16x32_bf16(
                    false, af[m], false, buf_[n], (short)0, accu[m][n], false, false);
            }
        __syncthreads();
    }

    // ---- epilogue: h = silu(g) * u, stored bf16 ----
    int mlane = (lane >= 16) ? 8 : 0;
    int ncol  = lane & 15;
#pragma unroll
    for (int m = 0; m < 2; ++m)
#pragma unroll
        for (int n = 0; n < 2; ++n) {
            int fcol = blockN * BN + wn * 32 + n * 16 + ncol;
#pragma unroll
            for (int r = 0; r < 8; ++r) {
                int c   = blockM * BM + wm * 32 + m * 16 + r + mlane;
                float g = accg[m][n][r];
                float u = accu[m][n][r];
                float hv = (g / (1.f + __expf(-g))) * u;
                H[(size_t)c * F + fcol] = (__bf16)hv;
            }
        }
}

// ================= GEMM2: y[token] += score * (H @ Wd^T) =============================
__global__ __launch_bounds__(256)
void gemm2_kernel(const __bf16* __restrict__ H, const float* __restrict__ Wd,
                  const int* __restrict__ ids, int base, const float* __restrict__ tsc,
                  float* __restrict__ Y, int K) {
    __shared__ __bf16 As[BM * LDH];
    __shared__ __bf16 Bs[BN * LDH];

    int tid  = threadIdx.x;
    int lane = tid & 31;
    int wid  = tid >> 5;
    int wm   = wid & 3;
    int wn   = wid >> 2;
    int blockN = blockIdx.x;   // DMODEL/64
    int blockM = blockIdx.y;   // CAP/128

    int arow  = tid >> 1;
    int ahalf = tid & 1;
    unsigned long long gA = (unsigned long long)(uintptr_t)
        (H + (size_t)(blockM * BM + arow) * K + ahalf * 16);
    unsigned ldsA = (unsigned)(uintptr_t)&As[arow * LDH + ahalf * 16];

    int brow = tid >> 2;
    int bq   = tid & 3;
    const float* bptr = Wd + (size_t)(blockN * BN + brow) * K + bq * 8;

    v8f acc[2][2];
#pragma unroll
    for (int m = 0; m < 2; ++m)
#pragma unroll
        for (int n = 0; n < 2; ++n) acc[m][n] = zero_v8f();

    for (int k0 = 0; k0 < K; k0 += BK) {
        async_copy_b128(ldsA, gA + (unsigned long long)k0 * 2);

        float4 b0 = *(const float4*)(bptr + k0);
        float4 b1 = *(const float4*)(bptr + k0 + 4);
        *(v8bf*)&Bs[brow * LDH + bq * 8] = cvt8_f32_bf16(b0, b1);

        if (k0 + BK < K) __builtin_prefetch(bptr + k0 + BK, 0, 1);
        wait_asynccnt0();
        __syncthreads();

        v16bf af[2], bf[2];
#pragma unroll
        for (int m = 0; m < 2; ++m) af[m] = load_frag_a(As, wm * 32 + m * 16, lane);
#pragma unroll
        for (int n = 0; n < 2; ++n) bf[n] = load_frag_b(Bs, wn * 32 + n * 16, lane);
#pragma unroll
        for (int m = 0; m < 2; ++m)
#pragma unroll
            for (int n = 0; n < 2; ++n)
                acc[m][n] = __builtin_amdgcn_wmma_f32_16x16x32_bf16(
                    false, af[m], false, bf[n], (short)0, acc[m][n], false, false);
        __syncthreads();
    }

    int mlane = (lane >= 16) ? 8 : 0;
    int ncol  = lane & 15;
#pragma unroll
    for (int m = 0; m < 2; ++m)
#pragma unroll
        for (int r = 0; r < 8; ++r) {
            int c    = blockM * BM + wm * 32 + m * 16 + r + mlane;
            int tok  = ids ? ids[c] : (base + c);
            float sc = tsc ? tsc[c] : 1.f;
            float* yr = Y + (size_t)tok * DMODEL;
#pragma unroll
            for (int n = 0; n < 2; ++n) {
                int fcol = blockN * BN + wn * 32 + n * 16 + ncol;
                atomicAdd(&yr[fcol], acc[m][n][r] * sc);
            }
        }
}

// ===================================== host ==========================================
extern "C" void kernel_launch(void* const* d_in, const int* in_sizes, int n_in,
                              void* d_out, int out_size, void* d_ws, size_t ws_size,
                              hipStream_t stream) {
    const float* x    = (const float*)d_in[0];
    const float* gw   = (const float*)d_in[1];
    const float* Wg   = (const float*)d_in[2];
    const float* Wu   = (const float*)d_in[3];
    const float* Wd   = (const float*)d_in[4];
    const float* Ws_g = (const float*)d_in[5];
    const float* Ws_u = (const float*)d_in[6];
    const float* Ws_d = (const float*)d_in[7];
    float* y = (float*)d_out;

    char* w = (char*)d_ws;
    float* scores  = (float*)w; w += (size_t)NEXP * NTOK * 4;   // 2 MB
    int*   topkIds = (int*)w;   w += (size_t)NEXP * CAP * 4;    // 256 KB
    float* topkSc  = (float*)w; w += (size_t)NEXP * CAP * 4;    // 256 KB
    int*   hist    = (int*)w;   w += (size_t)NEXP * NBINS * 4;  // 128 KB
    int*   thr     = (int*)w;   w += (size_t)NEXP * 3 * 4 + 64; // pad to 16B multiple
    int*   cntHi   = (int*)w;   w += (size_t)NEXP * 4;
    int*   cntEq   = (int*)w;   w += (size_t)NEXP * 4;
    __bf16* xb     = (__bf16*)w; w += (size_t)NTOK * DMODEL * 2; // 48 MB
    __bf16* hbuf   = (__bf16*)w; w += (size_t)CAP * FEXP * 2;    // 24 MB

    hipMemsetAsync(y,     0, (size_t)NTOK * DMODEL * 4, stream);
    hipMemsetAsync(hist,  0, (size_t)NEXP * NBINS * 4, stream);
    hipMemsetAsync(cntHi, 0, (size_t)NEXP * 4, stream);
    hipMemsetAsync(cntEq, 0, (size_t)NEXP * 4, stream);

    gate_kernel<<<NTOK / 256, 256, 0, stream>>>(x, gw, scores, xb);
    hist_kernel<<<dim3(64, NEXP), 256, 0, stream>>>(scores, hist);
    thresh_kernel<<<NEXP, 32, 0, stream>>>(hist, thr);
    compact_kernel<<<dim3(64, NEXP), 256, 0, stream>>>(scores, thr, cntHi, cntEq,
                                                       topkIds, topkSc);

    // routed experts: gather -> (gate|up) GEMM -> down GEMM with scatter-add
    for (int e = 0; e < NEXP; ++e) {
        const float* wg = Wg + (size_t)e * FEXP * DMODEL;
        const float* wu = Wu + (size_t)e * FEXP * DMODEL;
        const float* wd = Wd + (size_t)e * DMODEL * FEXP;
        gemm1_kernel<<<dim3(FEXP / BN, CAP / BM), 256, 0, stream>>>(
            xb, topkIds + e * CAP, 0, wg, wu, hbuf, FEXP, DMODEL);
        gemm2_kernel<<<dim3(DMODEL / BN, CAP / BM), 256, 0, stream>>>(
            hbuf, wd, topkIds + e * CAP, 0, topkSc + e * CAP, y, FEXP);
    }

    // shared expert: identity gather, unit score, F = 1536, 8 chunks of CAP tokens
    for (int c0 = 0; c0 < NTOK; c0 += CAP) {
        gemm1_kernel<<<dim3(FSH / BN, CAP / BM), 256, 0, stream>>>(
            xb, nullptr, c0, Ws_g, Ws_u, hbuf, FSH, DMODEL);
        gemm2_kernel<<<dim3(DMODEL / BN, CAP / BM), 256, 0, stream>>>(
            hbuf, Ws_d, nullptr, c0, nullptr, y, FSH);
    }
}
